// SAVSS2D_42262478193182
// MI455X (gfx1250) — matrix operand order repaired and checked by reading, not compiled, not run on hardware
//
#include <hip/hip_runtime.h>
#include <hip/hip_bf16.h>
#include <cstdint>

#define B_  16
#define C_  256
#define H_  96
#define W_  96
#define G_  16
#define HW_ (H_*W_)
#define P_  (B_*HW_)           // 147456 pixels
#define NE_ ((size_t)P_*C_)    // 37,748,736 elements
#define CPG_ (C_/G_)           // 16 channels per group

typedef __attribute__((ext_vector_type(16))) __bf16 v16bf;
typedef __attribute__((ext_vector_type(8)))  float  v8f;

union BFrag { v16bf v; uint4 q[2]; unsigned short s[16]; };

__device__ __forceinline__ unsigned short f2bf(float f) {
  union { float f; unsigned int u; } v; v.f = f;
  unsigned int u = v.u;
  u += 0x7fffu + ((u >> 16) & 1u);       // round-to-nearest-even
  return (unsigned short)(u >> 16);
}
__device__ __forceinline__ float bf2f(unsigned short s) {
  union { float f; unsigned int u; } v; v.u = ((unsigned int)s) << 16;
  return v.f;
}
__device__ __forceinline__ float sigm(float x) { return 1.0f / (1.0f + __expf(-x)); }

// GroupNorm affine fold: xn = x*ga + be ; then SiLU
__device__ __forceinline__ void gn_coefs(const float* stats, const float* gamma,
                                         const float* beta, int b, int c,
                                         float& ga, float& be) {
  const float inv_cnt = 1.0f / (float)(CPG_ * HW_);
  int g = c >> 4;
  float s1 = stats[(b * G_ + g) * 2 + 0], s2 = stats[(b * G_ + g) * 2 + 1];
  float mean = s1 * inv_cnt;
  float rstd = rsqrtf(s2 * inv_cnt - mean * mean + 1e-5f);
  ga = rstd * gamma[c];
  be = beta[c] - mean * ga;
}

// ---------------------------------------------------------------------------
// NCHW f32 -> pixel-major [p][C] bf16 via LDS tile transpose (both sides coalesced)
// Tile: 64 pixels x 64 channels. Grid: (P/64, C/64).
// ---------------------------------------------------------------------------
__global__ void cvt_x_k(const float* __restrict__ x, unsigned short* __restrict__ Xb) {
  __shared__ unsigned short t[64 * 72];          // padded rows: [px][c], 9 KB
  const int tid = threadIdx.x;
  const int pb  = blockIdx.x * 64;               // 9216 % 64 == 0 -> single batch per tile
  const int cb  = blockIdx.y * 64;
  const int b   = pb / HW_;
  const int hwb = pb % HW_;
  // read: NCHW, 64 consecutive hw per channel row (coalesced float4)
#pragma unroll
  for (int it = 0; it < 4; ++it) {
    int e  = tid + it * 256;
    int cl = e >> 4;                             // 0..63
    int hq = e & 15;                             // 4 hw values each
    float4 v = *(const float4*)(x + (size_t)(b * C_ + cb + cl) * HW_ + hwb + hq * 4);
    float f[4] = { v.x, v.y, v.z, v.w };
#pragma unroll
    for (int j = 0; j < 4; ++j)
      t[(hq * 4 + j) * 72 + cl] = f2bf(f[j]);
  }
  __syncthreads();
  // write: pixel-major, 8 uint4 per pixel (coalesced b128)
#pragma unroll
  for (int it = 0; it < 2; ++it) {
    int e  = tid + it * 256;
    int px = e >> 3;
    int cq = e & 7;
    union { uint4 q; unsigned short s[8]; } u;
#pragma unroll
    for (int r = 0; r < 8; ++r) u.s[r] = t[px * 72 + cq * 8 + r];
    *(uint4*)(Xb + (size_t)(pb + px) * C_ + cb + cq * 8) = u.q;
  }
}

__global__ void cvt_w_k(const float* __restrict__ w0, const float* __restrict__ w1,
                        const float* __restrict__ w2, unsigned short* __restrict__ Wb3) {
  int idx = blockIdx.x * blockDim.x + threadIdx.x;
  if (idx >= 3 * C_ * C_) return;
  const float* src = (idx < C_*C_) ? w0 : ((idx < 2*C_*C_) ? w1 : w2);
  Wb3[idx] = f2bf(src[idx % (C_*C_)]);
}

// ---------------------------------------------------------------------------
// WMMA GEMM: Y[o,p] = sum_c W[o,c] * X[c,p]
// Block: 256 threads = 8 waves. Block tile: 128(M) x 64(N), K = 256 in LDS.
// Staging uses CDNA5 GLOBAL_LOAD_ASYNC_TO_LDS_B128 (ASYNCcnt), then each wave
// waits on its own asyncs (s_wait_asynccnt 0) before the block barrier.
// MODE 0/2: write bf16 raw + accumulate GN stats (sum, sumsq) per (b, group)
// MODE 1  : staged tile is normalized (GN1+SiLU) in LDS; epilogue computes
//           fused = 0.25 * scansum * sigmoid(val + bias)
// ---------------------------------------------------------------------------
template<int MODE>
__global__ void gemm_wmma_k(const unsigned short* __restrict__ Wb,   // [C][C] bf16
                            const unsigned short* __restrict__ Xb,   // [P][C] bf16
                            unsigned short* __restrict__ Yb,         // [P][C] bf16
                            const float* __restrict__ ssum,          // MODE1: [P][C] f32
                            const float* __restrict__ bias,          // MODE1: [C]
                            float* __restrict__ stats_out,           // MODE0/2: [B*G][2]
                            const float* __restrict__ stats_in,      // MODE1
                            const float* __restrict__ gamma,         // MODE1
                            const float* __restrict__ beta)          // MODE1
{
  extern __shared__ unsigned short ldsX[];          // 64 pixels x 256 ch = 32 KB
  const int tid   = threadIdx.x;
  const int lane  = tid & 31;
  const int wv    = tid >> 5;                       // wave id 0..7
  const int pbase = blockIdx.x * 64;                // 64-pixel tile (never crosses batch)
  const int mbase = blockIdx.y * 128 + wv * 16;     // 16-row M tile == one GN group
  const int nlane = lane & 15;
  const int hi    = lane >> 4;
  const int arow  = mbase + nlane;                  // A-fragment row = M channel

  __builtin_prefetch((const void*)(Wb + (size_t)arow * C_), 0, 3);

  // ---- Async DMA staging: 32 KB contiguous, 16B per lane per op ----
  {
    const unsigned long long gbase = (unsigned long long)(Xb + (size_t)pbase * C_);
    const unsigned ldsbase = (unsigned)(size_t)(void*)ldsX;  // LDS offset = addr[31:0]
#pragma unroll
    for (int it = 0; it < 8; ++it) {
      const unsigned e = (unsigned)tid + (unsigned)it * 256u;
      unsigned loff = ldsbase + e * 16u;
      unsigned long long gaddr = gbase + (unsigned long long)e * 16ull;
      asm volatile("global_load_async_to_lds_b128 %0, %1, off"
                   :: "v"(loff), "v"(gaddr) : "memory");
    }
    asm volatile("s_wait_asynccnt 0x0" ::: "memory");
  }

  if (MODE == 1) {
    // Normalize the staged tile in LDS: GN1 + SiLU. Each thread rewrites the
    // same 16B chunks its own asyncs filled; cross-wave visibility via barrier.
    const int b = pbase / HW_;
    const float inv_cnt = 1.0f / (float)(CPG_ * HW_);
#pragma unroll
    for (int it = 0; it < 8; ++it) {
      const int e = tid + it * 256;
      const int cbase = (e & 31) * 8;               // 8 consecutive channels, one group
      const int g = cbase >> 4;
      float s1 = stats_in[(b * G_ + g) * 2 + 0], s2 = stats_in[(b * G_ + g) * 2 + 1];
      float mean = s1 * inv_cnt;
      float rstd = rsqrtf(s2 * inv_cnt - mean * mean + 1e-5f);
      union { uint4 q; unsigned short s[8]; } u;
      u.q = ((uint4*)ldsX)[e];
#pragma unroll
      for (int r = 0; r < 8; ++r) {
        float xn = (bf2f(u.s[r]) - mean) * rstd * gamma[cbase + r] + beta[cbase + r];
        u.s[r] = f2bf(xn * sigm(xn));
      }
      ((uint4*)ldsX)[e] = u.q;
    }
  }
  __syncthreads();

  v8f acc[4];
#pragma unroll
  for (int n = 0; n < 4; ++n) acc[n] = (v8f){0.f,0.f,0.f,0.f,0.f,0.f,0.f,0.f};

#pragma unroll
  for (int ks = 0; ks < 8; ++ks) {
    const int kb = ks * 32;
    // A 16x32 bf16 fragment: lanes 0-15 rows M, K = kb+8*hi+[0..7] then +16
    BFrag a;
    const uint4* ap = (const uint4*)(Wb + (size_t)arow * C_ + kb + hi * 8);
    a.q[0] = ap[0];
    a.q[1] = ap[2];
#pragma unroll
    for (int ns = 0; ns < 4; ++ns) {
      // B 32x16 bf16 fragment: lane column n, K = kb+8*hi+[0..7] then +16
      BFrag b;
      const uint4* bp = (const uint4*)(ldsX + (ns * 16 + nlane) * C_ + kb + hi * 8);
      b.q[0] = bp[0];
      b.q[1] = bp[2];
      acc[ns] = __builtin_amdgcn_wmma_f32_16x16x32_bf16(
          false, a.v, false, b.v, (short)0, acc[ns], false, false);
    }
  }

  float s1 = 0.f, s2 = 0.f;
#pragma unroll
  for (int ns = 0; ns < 4; ++ns) {
    const int p  = pbase + ns * 16 + nlane;
    const int m0 = mbase + hi * 8;                  // 8 consecutive output channels
    union { uint4 q; unsigned short s[8]; } ov;
    if (MODE == 1) {
      const float4* sp = (const float4*)(ssum + (size_t)p * C_ + m0);
      float4 sa = sp[0], sb = sp[1];
      float sv[8] = { sa.x, sa.y, sa.z, sa.w, sb.x, sb.y, sb.z, sb.w };
#pragma unroll
      for (int r = 0; r < 8; ++r) {
        float g = sigm(acc[ns][r] + bias[m0 + r]);
        ov.s[r] = f2bf(0.25f * sv[r] * g);
      }
    } else {
#pragma unroll
      for (int r = 0; r < 8; ++r) {
        float v = acc[ns][r];
        s1 += v; s2 += v * v;
        ov.s[r] = f2bf(v);
      }
    }
    *(uint4*)(Yb + (size_t)p * C_ + m0) = ov.q;     // single b128 store per fragment
  }

  if (MODE != 1) {
    // wave32 reduction; whole wave tile is one (batch, group) cell
#pragma unroll
    for (int off = 16; off >= 1; off >>= 1) {
      s1 += __shfl_xor(s1, off, 32);
      s2 += __shfl_xor(s2, off, 32);
    }
    if (lane == 0) {
      const int b = pbase / HW_;
      const int g = mbase / CPG_;
      atomicAdd(&stats_out[(b * G_ + g) * 2 + 0], s1);
      atomicAdd(&stats_out[(b * G_ + g) * 2 + 1], s2);
    }
  }
}

// ---------------------------------------------------------------------------
// Final GN2+SiLU: pixel-major bf16 -> NCHW f32 via LDS tile transpose.
// Tile: 64 pixels x 64 channels. Grid: (P/64, C/64). Both sides coalesced.
// ---------------------------------------------------------------------------
__global__ void gn_silu_out_k(const unsigned short* __restrict__ in,
                              float* __restrict__ out,
                              const float* __restrict__ stats,
                              const float* __restrict__ gamma,
                              const float* __restrict__ beta) {
  __shared__ float t[64 * 65];                   // [c][px], padded, ~16.6 KB
  const int tid = threadIdx.x;
  const int pb  = blockIdx.x * 64;
  const int cb  = blockIdx.y * 64;
  const int b   = pb / HW_;
  const int hwb = pb % HW_;
  const float inv_cnt = 1.0f / (float)(CPG_ * HW_);
  // read: pixel-major uint4 (8 bf16), apply GN2+SiLU, transpose into LDS
#pragma unroll
  for (int it = 0; it < 2; ++it) {
    int e  = tid + it * 256;
    int px = e >> 3;
    int cq = e & 7;
    union { uint4 q; unsigned short s[8]; } u;
    u.q = *(const uint4*)(in + (size_t)(pb + px) * C_ + cb + cq * 8);
    int c0 = cb + cq * 8;                        // 8 consecutive channels, one group
    int g  = c0 >> 4;
    float s1 = stats[(b * G_ + g) * 2 + 0], s2 = stats[(b * G_ + g) * 2 + 1];
    float mean = s1 * inv_cnt;
    float rstd = rsqrtf(s2 * inv_cnt - mean * mean + 1e-5f);
#pragma unroll
    for (int r = 0; r < 8; ++r) {
      float xn = (bf2f(u.s[r]) - mean) * rstd * gamma[c0 + r] + beta[c0 + r];
      t[(cq * 8 + r) * 65 + px] = xn * sigm(xn);
    }
  }
  __syncthreads();
  // write: NCHW, 64 consecutive hw per channel (coalesced f32)
#pragma unroll
  for (int it = 0; it < 16; ++it) {
    int e  = tid + it * 256;
    int cl = e >> 6;
    int px = e & 63;
    out[(size_t)(b * C_ + cb + cl) * HW_ + hwb + px] = t[cl * 65 + px];
  }
}

// ---------------------------------------------------------------------------
// Directional scans over GN1+SiLU-normalized input (norm applied inline):
// h = a*h + b*x ; y = c*h + d*x  (h0 = 0). Thread per row, C fastest.
// scan_w writes lr then adds rl; scan_h adds tb then bt.
// ---------------------------------------------------------------------------
__global__ void scan_w_k(const unsigned short* __restrict__ xp, float* __restrict__ ssum,
                         const float* __restrict__ ap, const float* __restrict__ bp,
                         const float* __restrict__ cp, const float* __restrict__ dp,
                         const float* __restrict__ stats, const float* __restrict__ gamma,
                         const float* __restrict__ beta) {
  int tid = blockIdx.x * blockDim.x + threadIdx.x;
  if (tid >= B_ * H_ * C_) return;
  int c = tid & (C_ - 1);
  int bh = tid >> 8;
  int h = bh % H_, b = bh / H_;
  float ga, be;
  gn_coefs(stats, gamma, beta, b, c, ga, be);
  size_t base = ((size_t)b * HW_ + (size_t)h * W_) * C_ + c;
  {
    float a0 = sigm(ap[c]), b0 = bp[c], c0 = cp[c], d0 = dp[c];
    float hs = 0.f;
    for (int w = 0; w < W_; ++w) {
      float xn = bf2f(xp[base + (size_t)w * C_]) * ga + be;
      float x = xn * sigm(xn);
      hs = a0 * hs + b0 * x;
      ssum[base + (size_t)w * C_] = c0 * hs + d0 * x;
    }
  }
  {
    float a1 = sigm(ap[C_ + c]), b1 = bp[C_ + c], c1 = cp[C_ + c], d1 = dp[C_ + c];
    float hs = 0.f;
    for (int w = W_ - 1; w >= 0; --w) {
      float xn = bf2f(xp[base + (size_t)w * C_]) * ga + be;
      float x = xn * sigm(xn);
      hs = a1 * hs + b1 * x;
      ssum[base + (size_t)w * C_] += c1 * hs + d1 * x;
    }
  }
}

__global__ void scan_h_k(const unsigned short* __restrict__ xp, float* __restrict__ ssum,
                         const float* __restrict__ ap, const float* __restrict__ bp,
                         const float* __restrict__ cp, const float* __restrict__ dp,
                         const float* __restrict__ stats, const float* __restrict__ gamma,
                         const float* __restrict__ beta) {
  int tid = blockIdx.x * blockDim.x + threadIdx.x;
  if (tid >= B_ * W_ * C_) return;
  int c = tid & (C_ - 1);
  int bw = tid >> 8;
  int w = bw % W_, b = bw / W_;
  float ga, be;
  gn_coefs(stats, gamma, beta, b, c, ga, be);
  size_t base = ((size_t)b * HW_ + w) * C_ + c;
  const size_t stride = (size_t)W_ * C_;
  {
    float a2 = sigm(ap[2*C_+c]), b2 = bp[2*C_+c], c2 = cp[2*C_+c], d2 = dp[2*C_+c];
    float hs = 0.f;
    for (int h = 0; h < H_; ++h) {
      float xn = bf2f(xp[base + h * stride]) * ga + be;
      float x = xn * sigm(xn);
      hs = a2 * hs + b2 * x;
      ssum[base + h * stride] += c2 * hs + d2 * x;
    }
  }
  {
    float a3 = sigm(ap[3*C_+c]), b3 = bp[3*C_+c], c3 = cp[3*C_+c], d3 = dp[3*C_+c];
    float hs = 0.f;
    for (int h = H_ - 1; h >= 0; --h) {
      float xn = bf2f(xp[base + h * stride]) * ga + be;
      float x = xn * sigm(xn);
      hs = a3 * hs + b3 * x;
      ssum[base + h * stride] += c3 * hs + d3 * x;
    }
  }
}

// ---------------------------------------------------------------------------
// Depthwise 3x3 SAME conv, pixel-major bf16 -> bf16 (lane-coalesced along C)
// ---------------------------------------------------------------------------
__global__ void dwconv_k(const unsigned short* __restrict__ in,
                         unsigned short* __restrict__ out,
                         const float* __restrict__ wgt) {     // (C,1,3,3)
  size_t idx = (size_t)blockIdx.x * blockDim.x + threadIdx.x;
  if (idx >= NE_) return;
  int c = (int)(idx & (C_ - 1));
  size_t p = idx >> 8;
  int hw = (int)(p % HW_);
  int b  = (int)(p / HW_);
  int h = hw / W_, w = hw % W_;
  float s = 0.f;
#pragma unroll
  for (int kh = 0; kh < 3; ++kh) {
    int hh = h + kh - 1;
    if (hh < 0 || hh >= H_) continue;
#pragma unroll
    for (int kw = 0; kw < 3; ++kw) {
      int ww = w + kw - 1;
      if (ww < 0 || ww >= W_) continue;
      s += bf2f(in[((size_t)b * HW_ + (size_t)hh * W_ + ww) * C_ + c]) * wgt[c * 9 + kh * 3 + kw];
    }
  }
  out[idx] = f2bf(s);
}

// ---------------------------------------------------------------------------
extern "C" void kernel_launch(void* const* d_in, const int* in_sizes, int n_in,
                              void* d_out, int out_size, void* d_ws, size_t ws_size,
                              hipStream_t stream) {
  (void)in_sizes; (void)n_in; (void)out_size; (void)ws_size;
  const float* x      = (const float*)d_in[0];
  const float* in_w   = (const float*)d_in[1];
  const float* gn1_w  = (const float*)d_in[2];
  const float* gn1_b  = (const float*)d_in[3];
  const float* a_p    = (const float*)d_in[4];
  const float* b_p    = (const float*)d_in[5];
  const float* c_p    = (const float*)d_in[6];
  const float* d_p    = (const float*)d_in[7];
  const float* gate_w = (const float*)d_in[8];
  const float* gate_b = (const float*)d_in[9];
  const float* dw_w   = (const float*)d_in[10];
  const float* pw_w   = (const float*)d_in[11];
  const float* gn2_w  = (const float*)d_in[12];
  const float* gn2_b  = (const float*)d_in[13];

  char* ws = (char*)d_ws;
  size_t o = 0;
  unsigned short* Wb3   = (unsigned short*)(ws + o); o += (size_t)3 * C_ * C_ * 2;
  float*          stats = (float*)(ws + o);          o += 4096;
  unsigned short* Xb    = (unsigned short*)(ws + o); o += NE_ * 2;   // x bf16 -> fused
  unsigned short* XPpre = (unsigned short*)(ws + o); o += NE_ * 2;   // xp_pre -> dwout
  float*          ssum  = (float*)(ws + o);          o += NE_ * 4;   // scansum -> out_pre
  unsigned short* fused  = Xb;                   // x_bf16 dead after GEMM1
  unsigned short* dwout  = XPpre;                // xp_pre dead after gate GEMM
  unsigned short* outpre = (unsigned short*)ssum;// ssum dead after gate GEMM
  float* stats0 = stats;
  float* stats1 = stats + B_ * G_ * 2;

  hipMemsetAsync(stats, 0, 2 * B_ * G_ * 2 * sizeof(float), stream);

  const unsigned nblk = (unsigned)((NE_ + 255) / 256);
  dim3 tgrid(P_ / 64, C_ / 64);             // transpose tiles
  cvt_w_k<<<(3 * C_ * C_ + 255) / 256, 256, 0, stream>>>(in_w, gate_w, pw_w, Wb3);
  cvt_x_k<<<tgrid, 256, 0, stream>>>(x, Xb);

  dim3 ggrid(P_ / 64, 2);
  const size_t lds = (size_t)64 * C_ * 2;   // 32 KB

  // 1) xp_pre = in_w @ x  (+ GN1 stats)
  gemm_wmma_k<0><<<ggrid, 256, lds, stream>>>(Wb3, Xb, XPpre, nullptr, nullptr,
                                              stats0, nullptr, nullptr, nullptr);
  // 2) scansum = lr + rl + tb + bt over xp = SiLU(GN1(xp_pre)) applied inline
  scan_w_k<<<(B_ * H_ * C_) / 256, 256, 0, stream>>>(XPpre, ssum, a_p, b_p, c_p, d_p,
                                                     stats0, gn1_w, gn1_b);
  scan_h_k<<<(B_ * W_ * C_) / 256, 256, 0, stream>>>(XPpre, ssum, a_p, b_p, c_p, d_p,
                                                     stats0, gn1_w, gn1_b);
  // 3) fused = 0.25 * scansum * sigmoid(gate_w @ xp + gate_b); xp formed in LDS
  gemm_wmma_k<1><<<ggrid, 256, lds, stream>>>(Wb3 + C_ * C_, XPpre, fused, ssum, gate_b,
                                              nullptr, stats0, gn1_w, gn1_b);
  // 4) depthwise 3x3
  dwconv_k<<<nblk, 256, 0, stream>>>(fused, dwout, dw_w);
  // 5) out_pre = pw_w @ dw  (+ GN2 stats)
  gemm_wmma_k<2><<<ggrid, 256, lds, stream>>>(Wb3 + 2 * C_ * C_, dwout, outpre, nullptr,
                                              nullptr, stats1, nullptr, nullptr, nullptr);
  // 6) out = SiLU(GN2(out_pre)) -> f32 NCHW (LDS transpose, both sides coalesced)
  gn_silu_out_k<<<tgrid, 256, 0, stream>>>(outpre, (float*)d_out, stats1, gn2_w, gn2_b);
}
